// GAN_63041529971278
// MI455X (gfx1250) — matrix operand order repaired and checked by reading, not compiled, not run on hardware
//
#include <hip/hip_runtime.h>

#define NHID 128
#define HHID 64

typedef __attribute__((ext_vector_type(16))) _Float16 v16h;
typedef __attribute__((ext_vector_type(8)))  _Float16 v8h;
typedef __attribute__((ext_vector_type(4)))  _Float16 v4h;
typedef __attribute__((ext_vector_type(8)))  float    v8f;

union AF { v16h v; v8h q[2]; _Float16 h[16]; };
union CF { v8f  v; float f[8]; };

__device__ __forceinline__ void fadd_agent(float* p, float v) {
  // Device-scope relaxed f32 atomic add -> global_atomic_add_f32 (no return).
  __hip_atomic_fetch_add(p, v, __ATOMIC_RELAXED, __HIP_MEMORY_SCOPE_AGENT);
}

// ---------------------------------------------------------------------------
// Kernel 1: zero the sum accumulator (lives in d_out) and the count buffer.
// ---------------------------------------------------------------------------
__global__ void k_zero(float* __restrict__ sums, float* __restrict__ counts,
                       size_t n_sums, size_t n_counts) {
  size_t stride = (size_t)gridDim.x * blockDim.x;
  size_t total  = n_sums + n_counts;
  for (size_t i = (size_t)blockIdx.x * blockDim.x + threadIdx.x; i < total; i += stride) {
    if (i < n_sums) sums[i] = 0.0f;
    else            counts[i - n_sums] = 0.0f;
  }
}

// ---------------------------------------------------------------------------
// Kernel 2: edge scatter-add. One wave32 per edge; each lane moves a float4
// (coalesced global_load_b128 of the 512B source row) and atomically adds it
// into the destination row (global_atomic_add_f32). Lane 0 counts the edge.
// ---------------------------------------------------------------------------
__global__ void k_scatter(const float* __restrict__ x,
                          const int*   __restrict__ src,
                          const int*   __restrict__ dst,
                          float*       __restrict__ sums,
                          float*       __restrict__ counts,
                          int nedges) {
  int gid  = blockIdx.x * blockDim.x + threadIdx.x;
  int e    = gid >> 5;
  int lane = gid & 31;
  if (e >= nedges) return;
  int s = src[e];
  int d = dst[e];
  const float4 v = ((const float4*)(x + (size_t)s * NHID))[lane];
  float* o = sums + (size_t)d * NHID + lane * 4;
  fadd_agent(o + 0, v.x);
  fadd_agent(o + 1, v.y);
  fadd_agent(o + 2, v.z);
  fadd_agent(o + 3, v.w);
  if (lane == 0) fadd_agent(counts + d, 1.0f);
}

// ---------------------------------------------------------------------------
// Kernel 3: per-node MLP via v_wmma_f32_16x16x32_f16.
// Block = 128 threads = 4 waves; each wave owns one 16-row tile.
// Weights stored N-major (transposed) in LDS so B fragment K runs are contiguous
// per lane (same K split as A: lanes 0-15 K {0..7,16..23}, lanes 16-31 K {8..15,24..31}) and load
// as 2x ds_load_b128. A fragments (two 8-half runs per lane) also load as
// 2x ds_load_b128 from row-major staging.
// ---------------------------------------------------------------------------
__launch_bounds__(128)
__global__ void k_mlp(const float* __restrict__ sums,
                      const float* __restrict__ counts,
                      const float* __restrict__ noise,
                      const float* __restrict__ W1, const float* __restrict__ b1,
                      const float* __restrict__ W2, const float* __restrict__ b2,
                      float* __restrict__ out, int nnodes) {
  __shared__ _Float16 w1t[HHID * NHID];   // [n][k] : 64 x 128 (N-major of W1)
  __shared__ _Float16 w2t[NHID * HHID];   // [n][k] : 128 x 64 (N-major of W2)
  __shared__ float    b1s[HHID];
  __shared__ float    b2s[NHID];
  // Per-wave scratch: g tile (16x128 f16) first, overlaid by H tile (16x64
  // f16) after g is consumed into registers (same-wave DS ops are in-order).
  __shared__ _Float16 sbuf[4][16 * NHID];

  const int tid = threadIdx.x;

  // ---- stage transposed f16 weights + biases (block-wide, one-time) ----
  for (int i = tid; i < HHID * NHID; i += 128) {
    const int n = i >> 7, k = i & 127;            // w1t[n][k] over 64x128
    w1t[i] = (_Float16)W1[k * HHID + n];
  }
  for (int i = tid; i < NHID * HHID; i += 128) {
    const int n = i >> 6, k = i & 63;             // w2t[n][k] over 128x64
    w2t[i] = (_Float16)W2[k * NHID + n];
  }
  if (tid < HHID) b1s[tid] = b1[tid];
  if (tid < NHID) b2s[tid] = b2[tid];
  __syncthreads();

  const int wave = tid >> 5;
  const int lane = tid & 31;
  const int tile = blockIdx.x * 4 + wave;
  const int row0 = tile * 16;
  if (row0 >= nnodes) return;

  _Float16* g = sbuf[wave];

  // ---- stage g = sums/max(cnt,1) + noise as f16 (packed ds_store_b64) ----
  for (int r = 0; r < 16; ++r) {
    const int grow = row0 + r;
    const float inv = 1.0f / fmaxf(counts[grow], 1.0f);
    const float4 sv = ((const float4*)(sums  + (size_t)grow * NHID))[lane];
    const float4 nv = ((const float4*)(noise + (size_t)grow * NHID))[lane];
    v4h p;
    p.x = (_Float16)(sv.x * inv + nv.x);
    p.y = (_Float16)(sv.y * inv + nv.y);
    p.z = (_Float16)(sv.z * inv + nv.z);
    p.w = (_Float16)(sv.w * inv + nv.w);
    *(v4h*)&g[r * NHID + lane * 4] = p;
  }

  // Lane roles. A layout (16-bit 16x32): lane<16 -> row=lane, K {0..7,16..23};
  // lane>=16 -> row=lane-16, K {8..15,24..31}. B layout: lane<16 -> col=lane,
  // K {0..7,16..23}; lane>=16 -> col=lane-16, K {8..15,24..31} (same as A).
  const int row   = lane & 15;
  const int half  = lane >> 4;
  const int hbase = half * 8;     // A-fragment K sub-offset
  const int krun  = half * 16;    // unused; B loads use hbase too
  const int col   = row;

  // ---- gather all layer-1 A chunks into registers (g then reusable) ----
  AF a1[4];
#pragma unroll
  for (int kc = 0; kc < 4; ++kc) {
    const int kb = kc * 32;
    a1[kc].q[0] = *(const v8h*)&g[row * NHID + kb + hbase];
    a1[kc].q[1] = *(const v8h*)&g[row * NHID + kb + 16 + hbase];
  }

  _Float16* hb = sbuf[wave];      // overlay: H tile, 16x64 f16 row-major

  // ---- layer 1: H = relu(G @ W1 + b1), 4 N-tiles x 4 K-chunks ----
#pragma unroll
  for (int nt = 0; nt < 4; ++nt) {
    const int nb = nt * 16;
    CF c;
    const float bias = b1s[nb + col];
#pragma unroll
    for (int j = 0; j < 8; ++j) c.f[j] = bias;
#pragma unroll
    for (int kc = 0; kc < 4; ++kc) {
      const int kb = kc * 32;
      AF b;
      b.q[0] = *(const v8h*)&w1t[(nb + col) * NHID + kb + hbase];
      b.q[1] = *(const v8h*)&w1t[(nb + col) * NHID + kb + 16 + hbase];
      c.v = __builtin_amdgcn_wmma_f32_16x16x32_f16(
          false, a1[kc].v, false, b.v, (short)0, c.v, false, false);
    }
    // D layout: VGPR j -> M = j + 8*half, N = col. ReLU into H staging.
#pragma unroll
    for (int j = 0; j < 8; ++j) {
      const int M = j + 8 * half;
      hb[M * HHID + nb + col] = (_Float16)fmaxf(c.f[j], 0.0f);
    }
  }

  // ---- gather layer-2 A chunks (K = 64 -> 2 chunks) ----
  AF a2[2];
#pragma unroll
  for (int kc = 0; kc < 2; ++kc) {
    const int kb = kc * 32;
    a2[kc].q[0] = *(const v8h*)&hb[row * HHID + kb + hbase];
    a2[kc].q[1] = *(const v8h*)&hb[row * HHID + kb + 16 + hbase];
  }

  // ---- layer 2: out = relu(H @ W2 + b2), 8 N-tiles x 2 K-chunks ----
#pragma unroll
  for (int nt = 0; nt < 8; ++nt) {
    const int nb = nt * 16;
    CF c;
    const float bias = b2s[nb + col];
#pragma unroll
    for (int j = 0; j < 8; ++j) c.f[j] = bias;
#pragma unroll
    for (int kc = 0; kc < 2; ++kc) {
      const int kb = kc * 32;
      AF b;
      b.q[0] = *(const v8h*)&w2t[(nb + col) * HHID + kb + hbase];
      b.q[1] = *(const v8h*)&w2t[(nb + col) * HHID + kb + 16 + hbase];
      c.v = __builtin_amdgcn_wmma_f32_16x16x32_f16(
          false, a2[kc].v, false, b.v, (short)0, c.v, false, false);
    }
#pragma unroll
    for (int j = 0; j < 8; ++j) {
      const int M = j + 8 * half;
      out[(size_t)(row0 + M) * NHID + nb + col] = fmaxf(c.f[j], 0.0f);
    }
  }
}

// ---------------------------------------------------------------------------
extern "C" void kernel_launch(void* const* d_in, const int* in_sizes, int n_in,
                              void* d_out, int out_size, void* d_ws, size_t ws_size,
                              hipStream_t stream) {
  const float* x     = (const float*)d_in[0];
  const int*   ei    = (const int*)  d_in[1];   // (2, E): src row then dst row
  // d_in[2] = batch (unused by the reference computation)
  const float* noise = (const float*)d_in[3];
  const float* W1    = (const float*)d_in[4];
  const float* b1    = (const float*)d_in[5];
  const float* W2    = (const float*)d_in[6];
  const float* b2    = (const float*)d_in[7];

  const int N = in_sizes[0] / NHID;
  const int E = in_sizes[1] / 2;
  const int* src = ei;
  const int* dst = ei + E;

  float* sums   = (float*)d_out;   // accumulate scatter-sum directly in d_out
  float* counts = (float*)d_ws;    // N floats of scratch

  // 1) zero accumulators
  k_zero<<<2048, 256, 0, stream>>>(sums, counts, (size_t)N * NHID, (size_t)N);

  // 2) edge scatter: one wave per edge
  {
    const long threads = (long)E * 32;
    const int  block   = 256;
    const long grid    = (threads + block - 1) / block;
    k_scatter<<<(dim3)(unsigned)grid, block, 0, stream>>>(x, src, dst, sums, counts, E);
  }

  // 3) WMMA MLP: 16 rows per wave, 4 waves per block
  {
    const int tiles  = (N + 15) / 16;
    const int blocks = (tiles + 3) / 4;
    k_mlp<<<blocks, 128, 0, stream>>>(sums, counts, noise, W1, b1, W2, b2,
                                      (float*)d_out, N);
  }
}
